// DEQModel_11879879541116
// MI455X (gfx1250) — compile-verified, matrix operand order checked by reading, not run on hardware
//
#include <hip/hip_runtime.h>

typedef __attribute__((ext_vector_type(16))) _Float16 v16h;
typedef __attribute__((ext_vector_type(8)))  _Float16 v8h;
typedef __attribute__((ext_vector_type(8)))  float    v8f;

#define B_IMG   32
#define C_CH    64
#define H_SZ    56
#define W_SZ    56
#define HW_SZ   (H_SZ * W_SZ)          // 3136
#define M_TOT   (B_IMG * HW_SZ)        // 100352 rows of the implicit GEMM
#define NELEM   (M_TOT * C_CH)         // 6422528
#define NTILES  (M_TOT / 16)           // 6272 (exact)
#define LDSB_HALFS (9 * 2 * 4 * 32 * 16)  // 36864 f16 = all weights, B-frag order

// ---------------------------------------------------------------------------
// prep: xb = x + b folded into NHWC f32 ; z1 = tanh(xb) as NHWC f16
// (first DEQ application: conv(0) = b, so f(0,x) = tanh(x + b))
// ---------------------------------------------------------------------------
__global__ void deq_prep(const float* __restrict__ x,
                         const float* __restrict__ b,
                         float* __restrict__ xb,
                         _Float16* __restrict__ z0) {
  int j = blockIdx.x * blockDim.x + threadIdx.x;
  if (j >= NELEM) return;
  int c  = j & 63;
  int m  = j >> 6;
  int n  = m / HW_SZ;
  int hw = m - n * HW_SZ;
  float v = x[(n * C_CH + c) * HW_SZ + hw] + b[c];
  xb[j] = v;
  z0[j] = (_Float16)tanhf(v);
}

// ---------------------------------------------------------------------------
// one DEQ iteration: z_out = tanh(conv3x3(z_in) + xb), implicit-GEMM WMMA
// One wave = 16 GEMM rows x all 64 output channels.
// ---------------------------------------------------------------------------
template <bool FINAL>
__global__ void deq_conv_iter(const _Float16* __restrict__ zin,   // NHWC f16
                              const float* __restrict__ xb,       // NHWC f32 (x+b)
                              const float* __restrict__ Wt,       // OIHW f32
                              _Float16* __restrict__ zout,        // NHWC f16
                              float* __restrict__ out_nchw) {     // NCHW f32
  extern __shared__ _Float16 ldsB[];   // 36864 halves = 73728 B

  // --- pack weights into B-fragment layout in LDS (once per workgroup) ---
  for (int idx = threadIdx.x; idx < LDSB_HALFS; idx += blockDim.x) {
    int e    = idx & 15;
    int lane = (idx >> 4) & 31;
    int nt   = (idx >> 9) & 3;      // N tile (co / 16)
    int k    = (idx >> 11) & 1;     // K chunk (ci / 32)
    int tap  = idx >> 12;           // 0..8
    int half = lane >> 4;
    int li   = lane & 15;
    int K  = ((e >= 8) ? 16 : 0) + half * 8 + (e & 7);
    int ci = k * 32 + K;
    int co = nt * 16 + li;
    ldsB[idx] = (_Float16)Wt[((co * C_CH + ci) * 3 + tap / 3) * 3 + (tap % 3)];
  }
  __syncthreads();

  int lane = threadIdx.x & 31;
  int wave = threadIdx.x >> 5;
  int tile = blockIdx.x * (blockDim.x >> 5) + wave;
  if (tile >= NTILES) return;

  int half = lane >> 4;
  int li   = lane & 15;
  int r    = tile * 16 + li;           // GEMM row handled by this lane
  int nImg = r / HW_SZ;
  int rem  = r - nImg * HW_SZ;
  int h    = rem / W_SZ;
  int w    = rem - h * W_SZ;

  v8f acc[4] = {};                     // 16x16 f32 accumulators for co tiles

  union Afrag { v16h v; v8h h2[2]; _Float16 s[16]; };

#pragma unroll
  for (int tap = 0; tap < 9; ++tap) {
    const int dh = tap / 3 - 1, dw = tap % 3 - 1;
    const int hh = h + dh, ww = w + dw;
    const bool valid = (hh >= 0) & (hh < H_SZ) & (ww >= 0) & (ww < W_SZ);
    const _Float16* src = zin + (size_t)(nImg * HW_SZ + hh * W_SZ + ww) * C_CH;

#pragma unroll
    for (int k = 0; k < 2; ++k) {
      Afrag a;
      if (valid) {
        // A layout (16-bit, 16x32): lane's K = {half*8+0..7} U {16+half*8+0..7}
        a.h2[0] = *(const v8h*)(src + k * 32 + half * 8);
        a.h2[1] = *(const v8h*)(src + k * 32 + 16 + half * 8);
      } else {
#pragma unroll
        for (int e = 0; e < 16; ++e) a.s[e] = (_Float16)0.f;  // SAME padding
      }
#pragma unroll
      for (int nt = 0; nt < 4; ++nt) {
        v16h bf = *(const v16h*)(ldsB + ((((tap * 2 + k) * 4 + nt) * 32 + lane) * 16));
        acc[nt] = __builtin_amdgcn_wmma_f32_16x16x32_f16(
            false, a.v, false, bf, (short)0, acc[nt], false, false);
      }
    }
  }

  // --- epilogue: + (x+b), tanh, store ---
#pragma unroll
  for (int nt = 0; nt < 4; ++nt) {
#pragma unroll
    for (int j = 0; j < 8; ++j) {
      int row = tile * 16 + j + half * 8;   // D VGPR j: M=j (lo half) / j+8 (hi half)
      int col = nt * 16 + li;
      float v = tanhf(acc[nt][j] + xb[row * C_CH + col]);
      if (!FINAL) {
        zout[row * C_CH + col] = (_Float16)v;
      } else {
        int ni = row / HW_SZ;
        int hw = row - ni * HW_SZ;
        out_nchw[(ni * C_CH + col) * HW_SZ + hw] = v;   // NCHW f32
      }
    }
  }
}

// ---------------------------------------------------------------------------
extern "C" void kernel_launch(void* const* d_in, const int* in_sizes, int n_in,
                              void* d_out, int out_size, void* d_ws, size_t ws_size,
                              hipStream_t stream) {
  const float* x  = (const float*)d_in[0];
  const float* Wt = (const float*)d_in[1];
  const float* b  = (const float*)d_in[2];
  float* out = (float*)d_out;

  // workspace layout (all 256B aligned): xb f32 | z ping | z pong  (~51.4 MB)
  char* ws = (char*)d_ws;
  float*    xb = (float*)ws;
  _Float16* zA = (_Float16*)(ws + (size_t)NELEM * 4);
  _Float16* zB = (_Float16*)(ws + (size_t)NELEM * 4 + (size_t)NELEM * 2);
  _Float16* zb[2] = {zA, zB};

  deq_prep<<<(NELEM + 255) / 256, 256, 0, stream>>>(x, b, xb, zA);

  const int NB = NTILES / 8;                  // 784 blocks x 8 waves = 6272 tiles
  const size_t ldsBytes = (size_t)LDSB_HALFS * 2;
  for (int it = 0; it < 24; ++it) {           // applications 2..25
    const _Float16* zi = zb[it & 1];
    _Float16*       zo = zb[(it + 1) & 1];
    if (it < 23)
      deq_conv_iter<false><<<NB, 256, ldsBytes, stream>>>(zi, xb, Wt, zo, nullptr);
    else
      deq_conv_iter<true><<<NB, 256, ldsBytes, stream>>>(zi, xb, Wt, nullptr, out);
  }
}
